// DPRNNLayer_89489938579879
// MI455X (gfx1250) — matrix operand order repaired and checked by reading, not compiled, not run on hardware
//
#include <hip/hip_runtime.h>
#include <hip/hip_bf16.h>

#define T_DIM 2048
#define B_DIM 64
#define I_DIM 256
#define H_DIM 256

typedef __attribute__((ext_vector_type(16))) __bf16 v16bf;
typedef __attribute__((ext_vector_type(8)))  __bf16 v8bf;
typedef __attribute__((ext_vector_type(8)))  float  v8f;
typedef __attribute__((ext_vector_type(4)))  float  v4f;
typedef __attribute__((ext_vector_type(4)))  unsigned int u32x4;
typedef __attribute__((ext_vector_type(8)))  int    i32x8;
typedef __attribute__((ext_vector_type(4)))  int    i32x4;

// Split a float into bf16 hi + bf16 lo (hi = rne(f), lo = rne(f - hi)).
static __device__ __forceinline__ void split4(v4f v, v16bf& hi, v16bf& lo, int base) {
#pragma unroll
    for (int j = 0; j < 4; ++j) {
        float f = v[j];
        __bf16 h = (__bf16)f;
        hi[base + j] = h;
        lo[base + j] = (__bf16)(f - (float)h);
    }
}

// Load 8 consecutive floats (16B-aligned) -> fragment elements [base, base+8).
static __device__ __forceinline__ void split8(const float* __restrict__ p,
                                              v16bf& hi, v16bf& lo, int base) {
    v4f v0 = *(const v4f*)(p);
    v4f v1 = *(const v4f*)(p + 4);
    split4(v0, hi, lo, base);
    split4(v1, hi, lo, base + 4);
}

static __device__ __forceinline__ v8f wmma_bf16(v16bf a, v16bf b, v8f c) {
    return __builtin_amdgcn_wmma_f32_16x16x32_bf16(
        /*neg_a=*/false, a, /*neg_b=*/false, b,
        /*c_mod=*/(short)0, c, /*reuse_a=*/false, /*reuse_b=*/false);
}

// ---------------------------------------------------------------------------
// TDM descriptor (CDNA5 ISA): DMA one contiguous row of `nelems`
// f32 from global to LDS. 1-row 2D tile => groups 2/3 (and the trailing
// group of the 6-arg builtin form) unused -> zero.
// ---------------------------------------------------------------------------
static __device__ __forceinline__ void tdm_load_row_f32(const float* gptr,
                                                        unsigned lds_byte_off,
                                                        unsigned nelems) {
    unsigned long long ga = (unsigned long long)(uintptr_t)gptr;
    u32x4 g0;
    g0[0] = 1u;                                    // count=1, is_restore=0, no gather
    g0[1] = lds_byte_off;                          // lds_addr (bytes)
    g0[2] = (unsigned)(ga & 0xffffffffull);        // global_addr[31:0]
    g0[3] = (unsigned)((ga >> 32) & 0x1ffffffull)  // global_addr[56:32]
          | (2u << 30);                            // type = 2 ("image")
    i32x8 g1;
    g1[0] = (int)(2u << 16);                       // wg_mask=0, data_size=2 (4B)
    g1[1] = (int)((nelems & 0xffffu) << 16);       // tensor_dim0[15:0]  (bits 63:48)
    g1[2] = (int)(((nelems >> 16) & 0xffffu)       // tensor_dim0[31:16] (bits 79:64)
          | (1u << 16));                           // tensor_dim1 = 1    (bits 95:80)
    g1[3] = (int)((nelems & 0xffffu) << 16);       // tile_dim0 = nelems (bits 127:112)
    g1[4] = 1;                                     // tile_dim1 = 1, tile_dim2 = 0
    g1[5] = (int)nelems;                           // tensor_dim0_stride[31:0]
    g1[6] = 0;
    g1[7] = 0;
    i32x4 gz4 = {0, 0, 0, 0};
    i32x8 gz8 = {0, 0, 0, 0, 0, 0, 0, 0};
    __builtin_amdgcn_tensor_load_to_lds(g0, g1, gz4, gz4, gz8, 0);
}

// ---------------------------------------------------------------------------
// Phase 1: x_proj[t,b,:] = x[t,b,:] @ w_ih^T + b_ih    (memory-bound GEMM)
// One wave per 16-row stripe of the (T*B) x H output; A fragments (x rows,
// split hi/lo) stay in registers across all 16 N-tiles; B fragments stream
// from L2-resident w_ih (256 KB).
// ---------------------------------------------------------------------------
__global__ __launch_bounds__(256)
void dprnn_xproj(const float* __restrict__ x, const float* __restrict__ w_ih,
                 const float* __restrict__ b_ih, float* __restrict__ out) {
    const int lane = threadIdx.x & 31;
    const int wave = threadIdx.x >> 5;
    const int m    = lane & 15;   // A row within tile
    const int g    = lane >> 4;   // lane half (selects K sub-block)
    const long stripe = (long)blockIdx.x * 8 + wave;   // 0..8191
    const long row0   = stripe * 16;                   // row in [T*B]

    const float* xrow = x + (row0 + m) * (long)I_DIM;

    // A fragments for the whole K=256 reduction (8 k-steps), hi/lo split.
    v16bf a_hi[8], a_lo[8];
#pragma unroll
    for (int ks = 0; ks < 8; ++ks) {
        const int kb = ks * 32;
        split8(xrow + kb + 8 * g,      a_hi[ks], a_lo[ks], 0);  // e<8 : K=kb+8g+e
        split8(xrow + kb + 16 + 8 * g, a_hi[ks], a_lo[ks], 8);  // e>=8: K=kb+16+8g+(e-8)
    }

    const int n = lane & 15;      // B/C column within tile
    for (int nt = 0; nt < 16; ++nt) {
        v8f acc = {0.f, 0.f, 0.f, 0.f, 0.f, 0.f, 0.f, 0.f};
        const float* wrowbase = w_ih + (nt * 16 + n) * I_DIM;
#pragma unroll
        for (int ks = 0; ks < 8; ++ks) {
            v16bf b_hi, b_lo;
            // B element e<8 -> K = ks*32 + 8g + e; e>=8 -> K = ks*32 + 16 + 8g + (e-8)
            split8(wrowbase + ks * 32 + 8 * g,      b_hi, b_lo, 0);
            split8(wrowbase + ks * 32 + 16 + 8 * g, b_hi, b_lo, 8);
            acc = wmma_bf16(a_hi[ks], b_hi, acc);
            acc = wmma_bf16(a_lo[ks], b_hi, acc);
            acc = wmma_bf16(a_hi[ks], b_lo, acc);
        }
        const float bv = b_ih[nt * 16 + n];
#pragma unroll
        for (int r = 0; r < 8; ++r) {                 // C: M = r + 8g, N = n
            const long mm = row0 + r + 8 * g;
            out[mm * (long)H_DIM + nt * 16 + n] = acc[r] + bv;
        }
    }
}

// ---------------------------------------------------------------------------
// Phase 2: persistent single-workgroup scan over T.
// 16 waves; wave w owns output column tile nt=w (all 4 row tiles).
// w_hh^T fragments live split-bf16 in VGPRs for the whole kernel; h lives
// split-bf16 in LDS (64 KB). x_proj[t+1] is DMA'd by the Tensor Data Mover
// into a double-buffered LDS stage (2 x 64 KB) while step t's WMMAs run;
// s_wait_tensorcnt(<=1) before the first barrier guarantees step t's stage
// is resident. Two barriers per step (read old h / publish new h).
// ---------------------------------------------------------------------------
__global__ __launch_bounds__(512)
void dprnn_scan(const float* __restrict__ h0, const float* __restrict__ w_hh,
                const float* __restrict__ b_hh, float* __restrict__ y,
                float* __restrict__ h_last) {
    __shared__ __bf16 sh_hi[B_DIM * H_DIM];       // 32 KB
    __shared__ __bf16 sh_lo[B_DIM * H_DIM];       // 32 KB
    __shared__ float  stage[2][B_DIM * H_DIM];    // 128 KB xp double buffer

    const int lane = threadIdx.x & 31;
    const int nt   = threadIdx.x >> 5;        // wave id == column tile
    const int n    = lane & 15;
    const int g    = lane >> 4;

    // Persistent B fragments of w_hh^T: B[k][n] = w_hh[nt*16+n][k], hi/lo split.
    v16bf whi[8], wlo[8];
    {
        const float* wrowbase = w_hh + (nt * 16 + n) * H_DIM;
#pragma unroll
        for (int ks = 0; ks < 8; ++ks) {
            split8(wrowbase + ks * 32 + 8 * g,      whi[ks], wlo[ks], 0);
            split8(wrowbase + ks * 32 + 16 + 8 * g, whi[ks], wlo[ks], 8);
        }
    }
    const float bias = b_hh[nt * 16 + n];

    const unsigned stage_off0 = (unsigned)(uintptr_t)&stage[0][0];
    const unsigned stage_off1 = (unsigned)(uintptr_t)&stage[1][0];

    // Kick off the TDM for step 0 while we seed h.
    if (nt == 0) {
        tdm_load_row_f32(y, stage_off0, B_DIM * H_DIM);
    }

    // Seed h from h0.
    for (int i = threadIdx.x; i < B_DIM * H_DIM; i += 512) {
        float f = h0[i];
        __bf16 h = (__bf16)f;
        sh_hi[i] = h;
        sh_lo[i] = (__bf16)(f - (float)h);
    }
    __syncthreads();

    for (int t = 0; t < T_DIM; ++t) {
        float* yrow = y + (long)t * (B_DIM * H_DIM);

        // Overlap: DMA next step's x_proj row into the other stage buffer.
        if (nt == 0 && t + 1 < T_DIM) {
            tdm_load_row_f32(yrow + B_DIM * H_DIM,
                             ((t + 1) & 1) ? stage_off1 : stage_off0,
                             B_DIM * H_DIM);
        }

        v8f acc[4];
#pragma unroll
        for (int mt = 0; mt < 4; ++mt) {
            v8f c = {0.f, 0.f, 0.f, 0.f, 0.f, 0.f, 0.f, 0.f};
            const __bf16* ph = sh_hi + (mt * 16 + (lane & 15)) * H_DIM;
            const __bf16* pl = sh_lo + (mt * 16 + (lane & 15)) * H_DIM;
#pragma unroll
            for (int ks = 0; ks < 8; ++ks) {
                const int kb = ks * 32;
                v8bf h0v = *(const v8bf*)(ph + kb + 8 * g);        // e<8
                v8bf h1v = *(const v8bf*)(ph + kb + 16 + 8 * g);   // e>=8
                v8bf l0v = *(const v8bf*)(pl + kb + 8 * g);
                v8bf l1v = *(const v8bf*)(pl + kb + 16 + 8 * g);
                v16bf ahi = __builtin_shufflevector(h0v, h1v,
                    0,1,2,3,4,5,6,7,8,9,10,11,12,13,14,15);
                v16bf alo = __builtin_shufflevector(l0v, l1v,
                    0,1,2,3,4,5,6,7,8,9,10,11,12,13,14,15);
                c = wmma_bf16(ahi, whi[ks], c);
                c = wmma_bf16(alo, whi[ks], c);
                c = wmma_bf16(ahi, wlo[ks], c);
            }
            acc[mt] = c;
        }

        // Oldest outstanding TDM descriptor == this step's stage buffer.
        if (nt == 0) {
            if (t + 1 < T_DIM) __builtin_amdgcn_s_wait_tensorcnt(1);
            else               __builtin_amdgcn_s_wait_tensorcnt(0);
        }
        __syncthreads();   // h reads done; staged xp(t) visible to all waves

        const float* xps = stage[t & 1];
#pragma unroll
        for (int mt = 0; mt < 4; ++mt) {
#pragma unroll
            for (int r = 0; r < 8; ++r) {
                const int b   = mt * 16 + r + 8 * g;   // C: M = r + 8g
                const int col = nt * 16 + n;
                float v  = acc[mt][r] + xps[b * H_DIM + col] + bias;
                float ht = tanhf(v);
                yrow[b * H_DIM + col] = ht;            // h_n[t]
                __bf16 h = (__bf16)ht;
                sh_hi[b * H_DIM + col] = h;
                sh_lo[b * H_DIM + col] = (__bf16)(ht - (float)h);
                if (t == T_DIM - 1) h_last[b * H_DIM + col] = ht;
            }
        }
        __syncthreads();   // new h published before next step reads it
    }
}

extern "C" void kernel_launch(void* const* d_in, const int* in_sizes, int n_in,
                              void* d_out, int out_size, void* d_ws, size_t ws_size,
                              hipStream_t stream) {
    (void)in_sizes; (void)n_in; (void)out_size; (void)d_ws; (void)ws_size;
    const float* x    = (const float*)d_in[0];
    const float* h0   = (const float*)d_in[1];
    const float* w_ih = (const float*)d_in[2];
    const float* b_ih = (const float*)d_in[3];
    const float* w_hh = (const float*)d_in[4];
    const float* b_hh = (const float*)d_in[5];

    float* y      = (float*)d_out;                          // h_n [T,B,H]
    float* h_last = y + (size_t)T_DIM * B_DIM * H_DIM;      // h_last [B,H]

    // Phase 1: 8192 stripes of 16 rows, 8 waves (256 threads) per block.
    dprnn_xproj<<<dim3((T_DIM * B_DIM) / (16 * 8)), dim3(256), 0, stream>>>(
        x, w_ih, b_ih, y);
    // Phase 2: one persistent workgroup (16 waves) does the sequential scan.
    dprnn_scan<<<dim3(1), dim3(512), 0, stream>>>(h0, w_hh, b_hh, y, h_last);
}
